// MultiHeadSelfAttention_30485677867264
// MI455X (gfx1250) — matrix operand order repaired and checked by reading, not compiled, not run on hardware
//
#include <hip/hip_runtime.h>
#include <hip/hip_bf16.h>
#include <hip/hip_fp16.h>
#include <cstdint>

// ---------------------------------------------------------------------------
// MHA forward for MI455X (gfx1250, wave32, WMMA f32<-f16).
//   x:[2,2048,1024] fp32; Wq/Wk/Wv/Wo:[1024,1024]; out:[2,2048,1024] fp32
// Pipeline: convert -> fused QKV WMMA GEMM -> flash attention -> out GEMM.
// Data movement: TDM tensor_load_to_lds (flash K/V tiles) and
// global_load_async_to_lds_b128 (GEMM tiles), with guarded fallbacks.
// ---------------------------------------------------------------------------

typedef __attribute__((ext_vector_type(16))) _Float16 v16h;
typedef __attribute__((ext_vector_type(8)))  _Float16 v8h;
typedef __attribute__((ext_vector_type(8)))  float    v8f;
typedef unsigned int u32x4 __attribute__((ext_vector_type(4)));
typedef int          i32x4 __attribute__((ext_vector_type(4)));
typedef int          i32x8 __attribute__((ext_vector_type(8)));

#define AS1 __attribute__((address_space(1)))
#define AS3 __attribute__((address_space(3)))

#define DMODEL 1024
#define NHEADS 16
#define DEPTH  64
#define BATCH  2
#define SEQ    2048
#define ROWS   (BATCH * SEQ)     // 4096

#if __has_builtin(__builtin_amdgcn_tensor_load_to_lds)
#define HAVE_TDM 1
#endif
#if defined(__has_include)
#if __has_include(<hip/amd_detail/amd_gfx1250_TDM.h>)
#define TDM_6ARG 1   // therock-10.0 headers -> 6-arg builtin
#endif
#endif

static __device__ __forceinline__ v8f wmma_f16(v16h a, v16h b, v8f c) {
  // D = A(16x32 f16) * B(32x16 f16) + C(16x16 f32)
  return __builtin_amdgcn_wmma_f32_16x16x32_f16(
      false, a, false, b, (short)0, c, false, false);
}

static __device__ __forceinline__ v8f zero8() {
  v8f z;
#pragma unroll
  for (int i = 0; i < 8; ++i) z[i] = 0.0f;
  return z;
}

static __device__ __forceinline__ v16h pack2(v8h lo, v8h hi) {
  v16h r;
#pragma unroll
  for (int t = 0; t < 8; ++t) { r[t] = lo[t]; r[t + 8] = hi[t]; }
  return r;
}

// A-fragment (16x32 f16): lane m = row; lanes 0-15 hold K=0..15, lanes 16-31 K=16..31
static __device__ __forceinline__ v16h a_frag(const _Float16* base, int m,
                                              int ld, int kbase, int hi8) {
  const _Float16* r = base + (size_t)m * ld + kbase;
  v8h lo = *(const v8h*)(r + 2 * hi8);
  v8h hh = *(const v8h*)(r + 2 * hi8 + 8);
  return pack2(lo, hh);
}

// B-fragment (32x16 f16): lanes 0-15 hold K=0..15 (col n=lane), lanes 16-31 K=16..31
static __device__ __forceinline__ v16h b_frag(const _Float16* base, int n,
                                              int ld, int koff) {
  const _Float16* r = base + (size_t)n * ld + koff;
  v8h lo = *(const v8h*)(r);
  v8h hh = *(const v8h*)(r + 8);
  return pack2(lo, hh);
}

// Cross-lane reductions within 16-lane halves (C-frag rows span 16 lanes).
template <int XM>
static __device__ __forceinline__ float swz_xor(float v) {
  int i = __builtin_bit_cast(int, v);
  i = __builtin_amdgcn_ds_swizzle(i, (XM << 10) | 0x1F);  // xor-mask mode
  return __builtin_bit_cast(float, i);
}
static __device__ __forceinline__ float redmax16(float v) {
  v = fmaxf(v, swz_xor<1>(v));
  v = fmaxf(v, swz_xor<2>(v));
  v = fmaxf(v, swz_xor<4>(v));
  v = fmaxf(v, swz_xor<8>(v));
  return v;
}
static __device__ __forceinline__ float redsum16(float v) {
  v += swz_xor<1>(v);
  v += swz_xor<2>(v);
  v += swz_xor<4>(v);
  v += swz_xor<8>(v);
  return v;
}

// --- CDNA5 async data movement helpers -------------------------------------
static __device__ __forceinline__ void async_b128(const _Float16* g, _Float16* l) {
#if __has_builtin(__builtin_amdgcn_global_load_async_to_lds_b128)
  __builtin_amdgcn_global_load_async_to_lds_b128(
      (AS1 i32x4*)(uintptr_t)g, (AS3 i32x4*)(uint32_t)(uintptr_t)l, 0, 0);
#else
  *(v8h*)l = *(const v8h*)g;   // synchronous fallback
#endif
}
static __device__ __forceinline__ void s_wait_async0() {
#if __has_builtin(__builtin_amdgcn_s_wait_asynccnt)
  __builtin_amdgcn_s_wait_asynccnt(0);
#else
  asm volatile("s_wait_asynccnt 0x0" ::: "memory");
#endif
}
static __device__ __forceinline__ void s_wait_tensor0() {
#if __has_builtin(__builtin_amdgcn_s_wait_tensorcnt)
  __builtin_amdgcn_s_wait_tensorcnt(0);
#else
  asm volatile("s_wait_tensorcnt 0x0" ::: "memory");
#endif
}

#ifdef HAVE_TDM
static __device__ __forceinline__ void tdm_load(u32x4 g0, i32x8 g1) {
  i32x4 z4 = {0, 0, 0, 0};
#ifdef TDM_6ARG
  i32x8 z8 = {0, 0, 0, 0, 0, 0, 0, 0};
  __builtin_amdgcn_tensor_load_to_lds(g0, g1, z4, z4, z8, 0);
#else
  __builtin_amdgcn_tensor_load_to_lds(g0, g1, z4, z4, 0);
#endif
}

// Build a 2D-tile D#: tile 64 rows x 64 halves (data_size=2B), LDS padded by
// 4 DWORDs every 32 DWORDs (row = 128B) -> LDS row stride 144B (72 halves).
static __device__ __forceinline__ void tdm_tile64x64(const _Float16* gsrc,
                                                     uint32_t lds_byte_addr,
                                                     int row_stride_halves) {
  const uint64_t ga = (uint64_t)(uintptr_t)gsrc;
  u32x4 g0;
  g0[0] = 1u;                                   // count=1, load, no gather
  g0[1] = lds_byte_addr;                        // lds_addr
  g0[2] = (uint32_t)ga;                         // global_addr[31:0]
  g0[3] = (uint32_t)((ga >> 32) & 0x01FFFFFFu)  // global_addr[56:32]
          | (2u << 30);                         // type=2 (image)
  i32x8 g1;
  g1[0] = (1 << 16)      // data_size = 2 bytes
        | (1 << 20)      // pad_enable
        | (4 << 22)      // pad_interval: 2<<4 = 32 DWORDs (one 128B row)
        | (3 << 25);     // pad_amount: 4 DWORDs (8 halves) -> stride 72
  g1[1] = (64 << 16);                    // tensor_dim0 = 64 (lo16 @ bits 63:48)
  g1[2] = (64 << 16);                    // tensor_dim0 hi=0; tensor_dim1 lo16 = 64
  g1[3] = (64 << 16);                    // tensor_dim1 hi=0; tile_dim0 = 64
  g1[4] = 64;                            // tile_dim1 = 64, tile_dim2 = 0
  g1[5] = row_stride_halves;             // tensor_dim0_stride lo32
  g1[6] = 0;
  g1[7] = 0;
  tdm_load(g0, g1);
}
#endif  // HAVE_TDM

// ---------------------------------------------------------------------------
// Kernel 0a: x fp32 -> f16
__global__ __launch_bounds__(256) void convert_x(const float* __restrict__ x,
                                                 _Float16* __restrict__ xh) {
  int idx = blockIdx.x * 256 + threadIdx.x;
  xh[idx] = (_Float16)x[idx];
}

// Kernel 0b: W[in][out] fp32 -> Wt[out][in] f16 (tiled transpose)
__global__ __launch_bounds__(256) void convert_wt(const float* __restrict__ W,
                                                  _Float16* __restrict__ Wt) {
  __shared__ float tile[32][33];
  const int bx = blockIdx.x, by = blockIdx.y;
  const int tx = threadIdx.x & 31, ty = threadIdx.x >> 5;
#pragma unroll
  for (int r = ty; r < 32; r += 8)
    tile[r][tx] = W[(size_t)(by * 32 + r) * DMODEL + bx * 32 + tx];
  __syncthreads();
#pragma unroll
  for (int r = ty; r < 32; r += 8)
    Wt[(size_t)(bx * 32 + r) * DMODEL + by * 32 + tx] = (_Float16)tile[tx][r];
}

// ---------------------------------------------------------------------------
// Kernel 1: fused QKV projection.  C[4096x3072] = xh[4096x1024] @ W + b
// Block tile 128x128, 8 waves in 2(M)x4(N); wave tile 64x32 (4x2 C-frags).
// Writes Q,K as [B,H,T,D] f16 and V transposed as [B,H,D,T] f16.
#define LDT 40   // padded LDS stride (halves) for 32-wide K tiles

__global__ __launch_bounds__(256) void gemm_qkv(
    const _Float16* __restrict__ xh, const _Float16* __restrict__ wt,
    const float* __restrict__ bq, const float* __restrict__ bk,
    const float* __restrict__ bv, _Float16* __restrict__ Qb,
    _Float16* __restrict__ Kb, _Float16* __restrict__ Vtb) {
  __shared__ _Float16 As[128 * LDT];
  __shared__ _Float16 Bs[128 * LDT];

  const int tid = threadIdx.x;
  const int lane = tid & 31, w = tid >> 5;
  const int mw = w >> 2, nw = w & 3;
  const int lm = lane & 15, hi8 = (lane >> 4) * 8;
  const int mt = blockIdx.y * 128;   // row tile in [0,4096)
  const int nt = blockIdx.x * 128;   // col tile in [0,3072)

  const int arow = tid >> 1, aseg = (tid & 1) * 16;

  v8f acc[4][2];
#pragma unroll
  for (int i = 0; i < 4; ++i)
#pragma unroll
    for (int j = 0; j < 2; ++j) acc[i][j] = zero8();

  for (int k0 = 0; k0 < DMODEL; k0 += 32) {
    __syncthreads();
    {  // async-stage tiles: 2x b128 per thread per tile (tracked by ASYNCcnt)
      const _Float16* ga = xh + (size_t)(mt + arow) * DMODEL + k0 + aseg;
      async_b128(ga,     As + arow * LDT + aseg);
      async_b128(ga + 8, As + arow * LDT + aseg + 8);
      const _Float16* gb = wt + (size_t)(nt + arow) * DMODEL + k0 + aseg;
      async_b128(gb,     Bs + arow * LDT + aseg);
      async_b128(gb + 8, Bs + arow * LDT + aseg + 8);
      // prefetch next K tile (global_prefetch_b8)
      __builtin_prefetch(ga + 32, 0, 1);
      __builtin_prefetch(gb + 32, 0, 1);
    }
    s_wait_async0();
    __syncthreads();

    v16h a[4], b[2];
#pragma unroll
    for (int i = 0; i < 4; ++i) a[i] = a_frag(As, mw * 64 + i * 16 + lm, LDT, 0, hi8);
#pragma unroll
    for (int j = 0; j < 2; ++j) b[j] = b_frag(Bs, nw * 32 + j * 16 + lm, LDT, (lane >> 4) * 16);
#pragma unroll
    for (int i = 0; i < 4; ++i)
#pragma unroll
      for (int j = 0; j < 2; ++j) acc[i][j] = wmma_f16(a[i], b[j], acc[i][j]);
  }

  // epilogue: bias + scatter into Q/K/Vt layouts
#pragma unroll
  for (int j = 0; j < 2; ++j) {
    const int n = nt + nw * 32 + j * 16 + lm;  // 16-wide frag never crosses 1024
#pragma unroll
    for (int i = 0; i < 4; ++i) {
      const int rbase = mt + mw * 64 + i * 16 + hi8;
      if (n < 1024) {
        const int h = n >> 6, d = n & 63;
        const float bias = bq[n];
#pragma unroll
        for (int v = 0; v < 8; ++v) {
          const int r = rbase + v, b = r >> 11, t = r & 2047;
          Qb[(((size_t)b * NHEADS + h) * SEQ + t) * DEPTH + d] =
              (_Float16)(acc[i][j][v] + bias);
        }
      } else if (n < 2048) {
        const int c = n - 1024, h = c >> 6, d = c & 63;
        const float bias = bk[c];
#pragma unroll
        for (int v = 0; v < 8; ++v) {
          const int r = rbase + v, b = r >> 11, t = r & 2047;
          Kb[(((size_t)b * NHEADS + h) * SEQ + t) * DEPTH + d] =
              (_Float16)(acc[i][j][v] + bias);
        }
      } else {
        const int c = n - 2048, h = c >> 6, d = c & 63;
        const float bias = bv[c];
#pragma unroll
        for (int v = 0; v < 8; ++v) {
          const int r = rbase + v, b = r >> 11, t = r & 2047;
          Vtb[(((size_t)b * NHEADS + h) * DEPTH + d) * SEQ + t] =
              (_Float16)(acc[i][j][v] + bias);
        }
      }
    }
  }
}

// ---------------------------------------------------------------------------
// Kernel 2: flash attention. Block = (b,h, 128 q rows); 8 waves x 16 q rows.
// K/V tiles (64 keys) staged by the Tensor Data Mover, shared by all 8 waves.
#define LDK 72   // padded stride for 64-wide key/d tiles (matches TDM pad)
#define LP  72   // padded stride for per-wave P transpose buffer

__global__ __launch_bounds__(256) void flash_attn(
    const _Float16* __restrict__ Qb, const _Float16* __restrict__ Kb,
    const _Float16* __restrict__ Vtb, _Float16* __restrict__ ctx) {
  __shared__ _Float16 Ks[64 * LDK];        // [key][d]
  __shared__ _Float16 Vs[64 * LDK];        // [d][token]
  __shared__ _Float16 Pl[8][16 * LP];      // per-wave P transpose staging

  const int tid = threadIdx.x;
  const int lane = tid & 31, w = tid >> 5;
  const int lm = lane & 15, hi8 = (lane >> 4) * 8;
  const int bh = blockIdx.y;               // 0..31
  const int qbase = blockIdx.x * 128 + w * 16;
  const int b = bh >> 4, h = bh & 15;

  const _Float16* Kbase = Kb + (size_t)bh * SEQ * DEPTH;
  const _Float16* Vbase = Vtb + (size_t)bh * DEPTH * SEQ;

  // Q fragments for this wave's 16 rows (two d-steps of 32)
  const _Float16* Qrow = Qb + (size_t)bh * SEQ * DEPTH;
  v16h aq[2];
#pragma unroll
  for (int s = 0; s < 2; ++s)
    aq[s] = a_frag(Qrow, qbase + lm, DEPTH, s * 32, hi8);

  v8f o[4];
#pragma unroll
  for (int j = 0; j < 4; ++j) o[j] = zero8();
  v8f mrow, lrow;
#pragma unroll
  for (int v = 0; v < 8; ++v) { mrow[v] = -3.0e38f; lrow[v] = 0.0f; }

  const float SC = 0.125f * 1.44269504f;   // 1/sqrt(64) in exp2 domain

#ifdef HAVE_TDM
  const uint32_t ldsK = (uint32_t)(uintptr_t)(void*)Ks;
  const uint32_t ldsV = (uint32_t)(uintptr_t)(void*)Vs;
#else
  const int trow = tid >> 2, tseg = (tid & 3) * 16;
#endif

  for (int kb = 0; kb < SEQ; kb += 64) {
    __syncthreads();                       // previous block fully consumed
#ifdef HAVE_TDM
    if (w == 0) {                          // one wave drives the TDM
      tdm_tile64x64(Kbase + (size_t)kb * DEPTH, ldsK, DEPTH);  // K: 64x64 dense
      tdm_tile64x64(Vbase + kb,              ldsV, SEQ);       // Vt rows, stride T
      s_wait_tensor0();
    }
#else
    {  // cooperative staging fallback
      const v8h* sk = (const v8h*)(Kbase + (size_t)(kb + trow) * DEPTH + tseg);
      v8h k0 = sk[0], k1 = sk[1];
      *(v8h*)(Ks + trow * LDK + tseg) = k0;
      *(v8h*)(Ks + trow * LDK + tseg + 8) = k1;
      const v8h* sv = (const v8h*)(Vbase + (size_t)trow * SEQ + kb + tseg);
      v8h v0 = sv[0], v1 = sv[1];
      *(v8h*)(Vs + trow * LDK + tseg) = v0;
      *(v8h*)(Vs + trow * LDK + tseg + 8) = v1;
    }
#endif
    __syncthreads();

    // S = Q @ K^T for 4 key chunks of 16 (each: 2 chained WMMAs over d)
    v8f s[4];
#pragma unroll
    for (int c = 0; c < 4; ++c) {
      v16h b0 = b_frag(Ks, c * 16 + lm, LDK, 0 + (lane >> 4) * 16);
      v16h b1 = b_frag(Ks, c * 16 + lm, LDK, 32 + (lane >> 4) * 16);
      v8f z = zero8();
      z = wmma_f16(aq[0], b0, z);
      z = wmma_f16(aq[1], b1, z);
      s[c] = z;
    }

    // online softmax (base-2 domain)
    v8f mn, al;
#pragma unroll
    for (int v = 0; v < 8; ++v) {
      float r = s[0][v] * SC;
      s[0][v] = r;
      float r1 = s[1][v] * SC; s[1][v] = r1; r = fmaxf(r, r1);
      float r2 = s[2][v] * SC; s[2][v] = r2; r = fmaxf(r, r2);
      float r3 = s[3][v] * SC; s[3][v] = r3; r = fmaxf(r, r3);
      r = redmax16(r);
      mn[v] = fmaxf(mrow[v], r);
      al[v] = exp2f(mrow[v] - mn[v]);
    }
#pragma unroll
    for (int c = 0; c < 4; ++c)
#pragma unroll
      for (int v = 0; v < 8; ++v) s[c][v] = exp2f(s[c][v] - mn[v]);
#pragma unroll
    for (int v = 0; v < 8; ++v) {
      float rs = redsum16(s[0][v] + s[1][v] + s[2][v] + s[3][v]);
      lrow[v] = lrow[v] * al[v] + rs;
      mrow[v] = mn[v];
    }

    // transpose P (C layout) -> A layout through per-wave LDS
    _Float16* pw = Pl[w];
#pragma unroll
    for (int c = 0; c < 4; ++c)
#pragma unroll
      for (int v = 0; v < 8; ++v)
        pw[(v + hi8) * LP + c * 16 + lm] = (_Float16)s[c][v];
    // same-wave DS ops are in-order: safe to read back without barrier
    v16h ap[2];
#pragma unroll
    for (int c2 = 0; c2 < 2; ++c2)
      ap[c2] = a_frag(pw, lm, LP, c2 * 32, hi8);

    // O = O*alpha + P @ V
#pragma unroll
    for (int j = 0; j < 4; ++j)
#pragma unroll
      for (int v = 0; v < 8; ++v) o[j][v] *= al[v];
#pragma unroll
    for (int c2 = 0; c2 < 2; ++c2)
#pragma unroll
      for (int j = 0; j < 4; ++j) {
        v16h bv = b_frag(Vs, j * 16 + lm, LDK, c2 * 32 + (lane >> 4) * 16);
        o[j] = wmma_f16(ap[c2], bv, o[j]);
      }
  }

  // normalize and write context [B,T,H*D] f16
#pragma unroll
  for (int j = 0; j < 4; ++j) {
    const int d = h * DEPTH + j * 16 + lm;
#pragma unroll
    for (int v = 0; v < 8; ++v) {
      const int t = qbase + hi8 + v;
      const float val = o[j][v] / lrow[v];
      ctx[((size_t)b * SEQ + t) * DMODEL + d] = (_Float16)val;
    }
  }
}

// ---------------------------------------------------------------------------
// Kernel 3: output projection. out[4096x1024] = ctx @ Wo + bo (fp32 out)
__global__ __launch_bounds__(256) void gemm_out(
    const _Float16* __restrict__ ch, const _Float16* __restrict__ wt,
    const float* __restrict__ bo, float* __restrict__ out) {
  __shared__ _Float16 As[128 * LDT];
  __shared__ _Float16 Bs[128 * LDT];

  const int tid = threadIdx.x;
  const int lane = tid & 31, w = tid >> 5;
  const int mw = w >> 2, nw = w & 3;
  const int lm = lane & 15, hi8 = (lane >> 4) * 8;
  const int mt = blockIdx.y * 128;
  const int nt = blockIdx.x * 128;
  const int arow = tid >> 1, aseg = (tid & 1) * 16;

  v8f acc[4][2];
#pragma unroll
  for (int i = 0; i < 4; ++i)
#pragma unroll
    for (int j = 0; j < 2; ++j) acc[i][j] = zero8();

  for (int k0 = 0; k0 < DMODEL; k0 += 32) {
    __syncthreads();
    {
      const _Float16* ga = ch + (size_t)(mt + arow) * DMODEL + k0 + aseg;
      async_b128(ga,     As + arow * LDT + aseg);
      async_b128(ga + 8, As + arow * LDT + aseg + 8);
      const _Float16* gb = wt + (size_t)(nt + arow) * DMODEL + k0 + aseg;
      async_b128(gb,     Bs + arow * LDT + aseg);
      async_b128(gb + 8, Bs + arow * LDT + aseg + 8);
      __builtin_prefetch(ga + 32, 0, 1);
      __builtin_prefetch(gb + 32, 0, 1);
    }
    s_wait_async0();
    __syncthreads();

    v16h a[4], b[2];
#pragma unroll
    for (int i = 0; i < 4; ++i) a[i] = a_frag(As, mw * 64 + i * 16 + lm, LDT, 0, hi8);
#pragma unroll
    for (int j = 0; j < 2; ++j) b[j] = b_frag(Bs, nw * 32 + j * 16 + lm, LDT, (lane >> 4) * 16);
#pragma unroll
    for (int i = 0; i < 4; ++i)
#pragma unroll
      for (int j = 0; j < 2; ++j) acc[i][j] = wmma_f16(a[i], b[j], acc[i][j]);
  }

#pragma unroll
  for (int j = 0; j < 2; ++j) {
    const int n = nt + nw * 32 + j * 16 + lm;
    const float bias = bo[n];
#pragma unroll
    for (int i = 0; i < 4; ++i) {
      const int rbase = mt + mw * 64 + i * 16 + hi8;
#pragma unroll
      for (int v = 0; v < 8; ++v)
        out[(size_t)(rbase + v) * DMODEL + n] = acc[i][j][v] + bias;
    }
  }
}

// ---------------------------------------------------------------------------
extern "C" void kernel_launch(void* const* d_in, const int* in_sizes, int n_in,
                              void* d_out, int out_size, void* d_ws,
                              size_t ws_size, hipStream_t stream) {
  (void)in_sizes; (void)n_in; (void)out_size; (void)ws_size;
  const float* x  = (const float*)d_in[0];
  const float* Wq = (const float*)d_in[1];
  const float* bq = (const float*)d_in[2];
  const float* Wk = (const float*)d_in[3];
  const float* bk = (const float*)d_in[4];
  const float* Wv = (const float*)d_in[5];
  const float* bv = (const float*)d_in[6];
  const float* Wo = (const float*)d_in[7];
  const float* bo = (const float*)d_in[8];
  float* out = (float*)d_out;

  _Float16* ws = (_Float16*)d_ws;
  size_t off = 0;
  _Float16* xh   = ws + off; off += (size_t)ROWS * DMODEL;          // 8 MB
  _Float16* wqkv = ws + off; off += (size_t)3 * DMODEL * DMODEL;    // 6 MB
  _Float16* wo   = ws + off; off += (size_t)DMODEL * DMODEL;        // 2 MB
  _Float16* Qb   = ws + off; off += (size_t)BATCH * NHEADS * SEQ * DEPTH;
  _Float16* Kb   = ws + off; off += (size_t)BATCH * NHEADS * SEQ * DEPTH;
  _Float16* Vtb  = ws + off; off += (size_t)BATCH * NHEADS * SEQ * DEPTH;
  _Float16* ctx  = ws + off; off += (size_t)ROWS * DMODEL;          // ~48 MB

  convert_x<<<(ROWS * DMODEL) / 256, 256, 0, stream>>>(x, xh);
  convert_wt<<<dim3(32, 32), 256, 0, stream>>>(Wq, wqkv);
  convert_wt<<<dim3(32, 32), 256, 0, stream>>>(Wk, wqkv + (size_t)DMODEL * DMODEL);
  convert_wt<<<dim3(32, 32), 256, 0, stream>>>(Wv, wqkv + (size_t)2 * DMODEL * DMODEL);
  convert_wt<<<dim3(32, 32), 256, 0, stream>>>(Wo, wo);

  gemm_qkv<<<dim3(24, 32), 256, 0, stream>>>(xh, wqkv, bq, bk, bv, Qb, Kb, Vtb);
  flash_attn<<<dim3(16, 32), 256, 0, stream>>>(Qb, Kb, Vtb, ctx);
  gemm_out<<<dim3(8, 32), 256, 0, stream>>>(ctx, wo, bo, out);
}
